// QuantAttention_86835648790565
// MI455X (gfx1250) — compile-verified, matrix operand order checked by reading, not run on hardware
//
#include <hip/hip_runtime.h>
#include <stdint.h>

// ---------------------------------------------------------------------------
// QuantAttention for MI455X (gfx1250): int8 fake-quant == exact iu8 WMMA path.
// Matrix math: V_WMMA_I32_16X16X64_IU8 (wave32).
// Tile staging: GLOBAL_LOAD_ASYNC_TO_LDS_B128 (ASYNCcnt) where no transform
// is needed; VGPR path only for the V transpose.
// ---------------------------------------------------------------------------

typedef __attribute__((ext_vector_type(8))) int v8i;
typedef signed char s8;

#define Bb    2
#define Nseq  2048
#define Dm    1024
#define Hh    16
#define Chd   64
#define Mrows (Bb * Nseq) // 4096

// scale slots (uint bits of absmax / minZ) at ws[0..63]
// 0:q 1:k 2:v 3:Wq 4:Wk 5:Wv 6:Wo 7:Qp 8:Kp 9:Vp 10:attn_out 11:minZ

__device__ __forceinline__ float scale_from(unsigned bits) {
  return __uint_as_float(bits) * (1.0f / 127.0f) + 1e-12f;
}

__device__ __forceinline__ v8i wmma_iu8(v8i a, v8i b, v8i c) {
  // signed A, signed B, int32 accumulate: D = A(16x64 i8) * B(64x16 i8) + C
  return __builtin_amdgcn_wmma_i32_16x16x64_iu8(true, a, true, b, c, false, false);
}

// Async 16B copy global -> LDS, tracked by ASYNCcnt (no VGPR round trip).
// Generic LDS pointer low 32 bits == LDS byte offset (ISA aperture rule).
__device__ __forceinline__ void async_ld_b128(void* lds, const void* gaddr) {
  unsigned loff = (unsigned)(unsigned long long)(uintptr_t)lds;
  asm volatile("global_load_async_to_lds_b128 %0, %1, off"
               :: "v"(loff), "v"(gaddr)
               : "memory");
}
__device__ __forceinline__ void wait_async0() {
  asm volatile("s_wait_asynccnt 0x0" ::: "memory");
}

// 16x64 int8 fragment from a row-major tile with 64B row stride.
// ISA 7.12.2 8-bit A layout: lane holds row (idxBase + lane%16); K-chunk base
// kb = (lane/16)*8; 8 bytes at K = kb + j*16 for j=0..3.
__device__ __forceinline__ v8i load_frag64(const s8* base, int idxBase, int lane) {
  int idx = idxBase + (lane & 15);
  int kb  = (lane >> 4) << 3;
  const s8* p = base + idx * 64 + kb;
  v8i f;
#pragma unroll
  for (int j = 0; j < 4; ++j) {
    const int* q = (const int*)(p + j * 16);
    f[2 * j]     = q[0];
    f[2 * j + 1] = q[1];
  }
  return f;
}

// Same fragment shape straight from global memory with arbitrary row stride.
__device__ __forceinline__ v8i load_frag_global(const s8* base, long rowStride,
                                                int rowBase, int lane) {
  int row = rowBase + (lane & 15);
  int kb  = (lane >> 4) << 3;
  const s8* p = base + (long)row * rowStride + kb;
  v8i f;
#pragma unroll
  for (int j = 0; j < 4; ++j) {
    const int* q = (const int*)(p + j * 16);
    f[2 * j]     = q[0];
    f[2 * j + 1] = q[1];
  }
  return f;
}

// reductions across the 16-lane half that owns one C-matrix row (wave32)
__device__ __forceinline__ float redmax16(float v) {
  v = fmaxf(v, __shfl_xor(v, 1, 32));
  v = fmaxf(v, __shfl_xor(v, 2, 32));
  v = fmaxf(v, __shfl_xor(v, 4, 32));
  v = fmaxf(v, __shfl_xor(v, 8, 32));
  return v;
}
__device__ __forceinline__ float redsum16(float v) {
  v += __shfl_xor(v, 1, 32);
  v += __shfl_xor(v, 2, 32);
  v += __shfl_xor(v, 4, 32);
  v += __shfl_xor(v, 8, 32);
  return v;
}

// ---------------------------------------------------------------------------
__global__ void qa_init_scales(unsigned* s) {
  int t = threadIdx.x;
  if (t < 16) s[t] = (t == 11) ? 0x7F800000u : 0u; // minZ = +inf
}

__global__ void qa_absmax(const float* __restrict__ x, long n, unsigned* slot) {
  long i = (long)blockIdx.x * blockDim.x + threadIdx.x;
  long stride = (long)gridDim.x * blockDim.x;
  float m = 0.0f;
  for (; i < n; i += stride) m = fmaxf(m, fabsf(x[i]));
#pragma unroll
  for (int d = 16; d; d >>= 1) m = fmaxf(m, __shfl_xor(m, d, 32));
  if ((threadIdx.x & 31) == 0) atomicMax(slot, __float_as_uint(m));
}

__global__ void qa_quantize(const float* __restrict__ x, s8* __restrict__ y,
                            long n, const unsigned* slot) {
  float s   = scale_from(*slot);
  float inv = 1.0f / s;
  long i = (long)blockIdx.x * blockDim.x + threadIdx.x;
  long stride = (long)gridDim.x * blockDim.x;
  for (; i < n; i += stride) {
    float v = fminf(fmaxf(x[i] * inv, -128.0f), 127.0f);
    y[i] = (s8)__float2int_rn(v);
  }
}

// ---------------------------------------------------------------------------
// C[M,N] = (A_i8[M,K] . B_i8[N,K]^T) * sA*sB + bias ; optional fused absmax(C)
// block tile 128x128, 8 waves (4x2), wave tile 32x64 = 2x4 WMMA tiles, K step 64
// ---------------------------------------------------------------------------
__global__ __launch_bounds__(256) void qa_gemm_iu8(
    const s8* __restrict__ A, const s8* __restrict__ Bw,
    const float* __restrict__ bias, float* __restrict__ C,
    int M, int Nn, int K,
    const unsigned* __restrict__ scales, int slotA, int slotB,
    unsigned* amaxOut) {
  __shared__ s8 lA[128 * 64]; // 8 KB
  __shared__ s8 lB[128 * 64]; // 8 KB
  const int tid = threadIdx.x, wave = tid >> 5, lane = tid & 31;
  const int bm = blockIdx.x * 128, bn = blockIdx.y * 128;
  const int wM = wave >> 1, wN = wave & 1;

  v8i acc[2][4] = {};

  for (int k0 = 0; k0 < K; k0 += 64) {
    __syncthreads();
    // stage 128x64 i8 A and B slabs via async LDS copies:
    // 512 x 16B chunks each, 2 per thread
#pragma unroll
    for (int i = 0; i < 2; ++i) {
      int c = tid * 2 + i;
      int r = c >> 2, o = (c & 3) * 16;
      async_ld_b128(&lA[r * 64 + o], A + (long)(bm + r) * K + k0 + o);
      async_ld_b128(&lB[r * 64 + o], Bw + (long)(bn + r) * K + k0 + o);
    }
    if (k0 + 64 < K) { // prefetch next slabs -> global_prefetch_b8
      int c = tid * 2;
      int r = c >> 2, o = (c & 3) * 16;
      __builtin_prefetch(A + (long)(bm + r) * K + (k0 + 64) + o, 0, 1);
      __builtin_prefetch(Bw + (long)(bn + r) * K + (k0 + 64) + o, 0, 1);
    }
    wait_async0();
    __syncthreads();

    v8i af[2], bf[4];
#pragma unroll
    for (int t = 0; t < 2; ++t) af[t] = load_frag64(lA, wM * 32 + t * 16, lane);
#pragma unroll
    for (int u = 0; u < 4; ++u) bf[u] = load_frag64(lB, wN * 64 + u * 16, lane);
#pragma unroll
    for (int t = 0; t < 2; ++t)
#pragma unroll
      for (int u = 0; u < 4; ++u) acc[t][u] = wmma_iu8(af[t], bf[u], acc[t][u]);
  }

  const float scale = scale_from(scales[slotA]) * scale_from(scales[slotB]);
  const int half8 = (lane >> 4) * 8, col0 = lane & 15;
  float am = 0.0f;
#pragma unroll
  for (int t = 0; t < 2; ++t)
#pragma unroll
    for (int u = 0; u < 4; ++u) {
      int colBase = bn + wN * 64 + u * 16 + col0;
      float bcol = bias[colBase];
#pragma unroll
      for (int r = 0; r < 8; ++r) {
        int row = bm + wM * 32 + t * 16 + r + half8;
        float v = (float)acc[t][u][r] * scale + bcol;
        C[(long)row * Nn + colBase] = v;
        am = fmaxf(am, fabsf(v));
      }
    }
  if (amaxOut) atomicMax(amaxOut, __float_as_uint(am));
}

// ---------------------------------------------------------------------------
// Attention pass A: per-row online-softmax stats (rowmax, Z) of logits
//   S = (Qp_i8 . Kp_i8^T) * sQ*sK / sqrt(64),  plus global atomicMin(Z).
// grid: (B*H, N/128); 8 waves x 16 query rows.
// ---------------------------------------------------------------------------
__global__ __launch_bounds__(256) void qa_attn_stats(
    const s8* __restrict__ Qq, const s8* __restrict__ Kq,
    const unsigned* __restrict__ scales,
    float* __restrict__ rowM, float* __restrict__ rowZ, unsigned* minZ) {
  __shared__ s8 lK[64 * 64]; // 64-key slab
  const int tid = threadIdx.x, wave = tid >> 5, lane = tid & 31;
  const int bh = blockIdx.x, b = bh >> 4, h = bh & 15;
  const int qBase = blockIdx.y * 128 + wave * 16;
  const int half8 = (lane >> 4) * 8;

  const s8* qhead = Qq + (long)b * Nseq * Dm + h * 64;
  const s8* khead = Kq + (long)b * Nseq * Dm + h * 64;
  v8i aq = load_frag_global(qhead, Dm, qBase, lane);

  const float sc = scale_from(scales[7]) * scale_from(scales[8]) * 0.125f; // /sqrt(64)

  float mrow[8], zrow[8];
#pragma unroll
  for (int r = 0; r < 8; ++r) { mrow[r] = -1e30f; zrow[r] = 0.0f; }

  for (int kc = 0; kc < Nseq; kc += 64) {
    __syncthreads();
    { // stage 64x64 i8 keys (async global -> LDS)
      int r = tid >> 2, o = (tid & 3) * 16;
      async_ld_b128(&lK[r * 64 + o], khead + (long)(kc + r) * Dm + o);
    }
    wait_async0();
    __syncthreads();
#pragma unroll
    for (int sub = 0; sub < 4; ++sub) {
      v8i z = {};
      v8i bk = load_frag64(lK, sub * 16, lane);
      v8i s  = wmma_iu8(aq, bk, z);
#pragma unroll
      for (int r = 0; r < 8; ++r) {
        float sv   = (float)s[r] * sc;
        float tmax = redmax16(sv);
        float mnew = fmaxf(mrow[r], tmax);
        float es   = redsum16(__expf(sv - mnew));
        zrow[r]    = zrow[r] * __expf(mrow[r] - mnew) + es;
        mrow[r]    = mnew;
      }
    }
  }
  if ((lane & 15) == 0) {
#pragma unroll
    for (int r = 0; r < 8; ++r) {
      long idx = (long)bh * Nseq + qBase + r + half8;
      rowM[idx] = mrow[r];
      rowZ[idx] = zrow[r];
      atomicMin(minZ, __float_as_uint(zrow[r])); // max prob = 1/minZ
    }
  }
}

// ---------------------------------------------------------------------------
// Attention pass B: recompute logits, softmax with saved stats, requantize
// probs to i8 (scale sP = (1/minZ)/127), PV via iu8 WMMA. Out fp32 + absmax.
// ---------------------------------------------------------------------------
__global__ __launch_bounds__(256) void qa_attn_pv(
    const s8* __restrict__ Qq, const s8* __restrict__ Kq, const s8* __restrict__ Vq,
    const unsigned* __restrict__ scales,
    const float* __restrict__ rowM, const float* __restrict__ rowZ,
    float* __restrict__ Out, unsigned* amaxOut) {
  __shared__ s8 lK[64 * 64];       // keys slab
  __shared__ s8 lVt[64 * 64];      // V slab transposed: [channel][key]
  __shared__ s8 lP[8 * 16 * 64];   // per-wave prob tile 16x64
  const int tid = threadIdx.x, wave = tid >> 5, lane = tid & 31;
  const int bh = blockIdx.x, b = bh >> 4, h = bh & 15;
  const int qBase = blockIdx.y * 128 + wave * 16;
  const int half8 = (lane >> 4) * 8, col0 = lane & 15;
  s8* myP = lP + wave * (16 * 64);

  const s8* qhead = Qq + (long)b * Nseq * Dm + h * 64;
  const s8* khead = Kq + (long)b * Nseq * Dm + h * 64;
  const s8* vhead = Vq + (long)b * Nseq * Dm + h * 64;
  v8i aq = load_frag_global(qhead, Dm, qBase, lane);

  const float sc    = scale_from(scales[7]) * scale_from(scales[8]) * 0.125f;
  const float sP    = (1.0f / __uint_as_float(scales[11])) * (1.0f / 127.0f) + 1e-12f;
  const float invSP = 1.0f / sP;
  const float sV    = scale_from(scales[9]);

  float mr[8], zinv[8];
#pragma unroll
  for (int r = 0; r < 8; ++r) {
    long idx = (long)bh * Nseq + qBase + r + half8;
    mr[r]   = rowM[idx];
    zinv[r] = 1.0f / rowZ[idx];
  }

  v8i pacc[4] = {};

  for (int kc = 0; kc < Nseq; kc += 64) {
    __syncthreads();
    { // stage keys async (row-major) and V via VGPR transpose (byte-scatter)
      int r = tid >> 2, o = (tid & 3) * 16;
      async_ld_b128(&lK[r * 64 + o], khead + (long)(kc + r) * Dm + o);
      uint4 vv = *(const uint4*)(vhead + (long)(kc + r) * Dm + o);
      unsigned w[4] = {vv.x, vv.y, vv.z, vv.w};
#pragma unroll
      for (int i = 0; i < 16; ++i)
        lVt[(o + i) * 64 + r] = (s8)((w[i >> 2] >> ((i & 3) * 8)) & 0xFF);
    }
    wait_async0();
    __syncthreads();

    // logits -> probs -> i8, packed via LDS into A-fragment layout
#pragma unroll
    for (int sub = 0; sub < 4; ++sub) {
      v8i z = {};
      v8i bk = load_frag64(lK, sub * 16, lane);
      v8i s  = wmma_iu8(aq, bk, z);
#pragma unroll
      for (int r = 0; r < 8; ++r) {
        float p  = __expf((float)s[r] * sc - mr[r]) * zinv[r];
        int   pq = __float2int_rn(fminf(p * invSP, 127.0f));
        myP[(r + half8) * 64 + sub * 16 + col0] = (s8)pq;
      }
    }
    __syncthreads();

    v8i pf = load_frag64(myP, 0, lane); // 16 q-rows x 64 keys
#pragma unroll
    for (int u = 0; u < 4; ++u) {
      v8i vf = load_frag64(lVt, u * 16, lane); // B: channel cols, K = keys
      pacc[u] = wmma_iu8(pf, vf, pacc[u]);
    }
  }

  const float so = sP * sV;
  float am = 0.0f;
#pragma unroll
  for (int u = 0; u < 4; ++u)
#pragma unroll
    for (int r = 0; r < 8; ++r) {
      int row = qBase + r + half8;
      int col = h * 64 + u * 16 + col0;
      float v = (float)pacc[u][r] * so;
      Out[(long)(b * Nseq + row) * Dm + col] = v;
      am = fmaxf(am, fabsf(v));
    }
  atomicMax(amaxOut, __float_as_uint(am));
}

// ---------------------------------------------------------------------------
extern "C" void kernel_launch(void* const* d_in, const int* in_sizes, int n_in,
                              void* d_out, int out_size, void* d_ws, size_t ws_size,
                              hipStream_t stream) {
  (void)in_sizes; (void)n_in; (void)out_size; (void)ws_size;
  const float* q  = (const float*)d_in[0];
  const float* k  = (const float*)d_in[1];
  const float* v  = (const float*)d_in[2];
  const float* Wq = (const float*)d_in[3];
  const float* bq = (const float*)d_in[4];
  const float* Wk = (const float*)d_in[5];
  const float* bk = (const float*)d_in[6];
  const float* Wv = (const float*)d_in[7];
  const float* bv = (const float*)d_in[8];
  const float* Wo = (const float*)d_in[9];
  const float* bo = (const float*)d_in[10];
  float* out = (float*)d_out;

  char* ws = (char*)d_ws;
  unsigned* scales = (unsigned*)ws;
  size_t off = 256;
  const size_t actI8 = (size_t)Mrows * Dm;      // 4 MiB
  const size_t wI8   = (size_t)Dm * Dm;         // 1 MiB
  const size_t actF  = actI8 * 4;               // 16 MiB
  const size_t rows  = (size_t)Bb * Hh * Nseq;  // 65536

  s8* qi8  = (s8*)(ws + off); off += actI8;
  s8* ki8  = (s8*)(ws + off); off += actI8;
  s8* vi8  = (s8*)(ws + off); off += actI8;
  s8* wqi8 = (s8*)(ws + off); off += wI8;
  s8* wki8 = (s8*)(ws + off); off += wI8;
  s8* wvi8 = (s8*)(ws + off); off += wI8;
  s8* woi8 = (s8*)(ws + off); off += wI8;
  float* Qp = (float*)(ws + off); off += actF;
  float* Kp = (float*)(ws + off); off += actF;
  float* Vp = (float*)(ws + off); off += actF;
  s8* Qpi8 = (s8*)(ws + off); off += actI8;
  s8* Kpi8 = (s8*)(ws + off); off += actI8;
  s8* Vpi8 = (s8*)(ws + off); off += actI8;
  float* rowMv = (float*)(ws + off); off += rows * 4;
  float* rowZv = (float*)(ws + off); off += rows * 4;
  float* Att   = (float*)(ws + off); off += actF;
  s8* atti8    = (s8*)(ws + off); off += actI8;

  const long nAct = (long)Mrows * Dm, nW = (long)Dm * Dm;
  dim3 blk(256);

  // 1) reset atomic scale slots (deterministic each call)
  qa_init_scales<<<1, 32, 0, stream>>>(scales);

  // 2) per-tensor absmax
  qa_absmax<<<512, blk, 0, stream>>>(q,  nAct, scales + 0);
  qa_absmax<<<512, blk, 0, stream>>>(k,  nAct, scales + 1);
  qa_absmax<<<512, blk, 0, stream>>>(v,  nAct, scales + 2);
  qa_absmax<<<256, blk, 0, stream>>>(Wq, nW,   scales + 3);
  qa_absmax<<<256, blk, 0, stream>>>(Wk, nW,   scales + 4);
  qa_absmax<<<256, blk, 0, stream>>>(Wv, nW,   scales + 5);
  qa_absmax<<<256, blk, 0, stream>>>(Wo, nW,   scales + 6);

  // 3) quantize inputs + weights
  qa_quantize<<<1024, blk, 0, stream>>>(q,  qi8,  nAct, scales + 0);
  qa_quantize<<<1024, blk, 0, stream>>>(k,  ki8,  nAct, scales + 1);
  qa_quantize<<<1024, blk, 0, stream>>>(v,  vi8,  nAct, scales + 2);
  qa_quantize<<<512,  blk, 0, stream>>>(Wq, wqi8, nW,   scales + 3);
  qa_quantize<<<512,  blk, 0, stream>>>(Wk, wki8, nW,   scales + 4);
  qa_quantize<<<512,  blk, 0, stream>>>(Wv, wvi8, nW,   scales + 5);
  qa_quantize<<<512,  blk, 0, stream>>>(Wo, woi8, nW,   scales + 6);

  // 4) projections (iu8 WMMA GEMMs) with fused absmax of outputs
  dim3 gGemm(Mrows / 128, Dm / 128);
  qa_gemm_iu8<<<gGemm, blk, 0, stream>>>(qi8, wqi8, bq, Qp, Mrows, Dm, Dm,
                                         scales, 0, 3, scales + 7);
  qa_gemm_iu8<<<gGemm, blk, 0, stream>>>(ki8, wki8, bk, Kp, Mrows, Dm, Dm,
                                         scales, 1, 4, scales + 8);
  qa_gemm_iu8<<<gGemm, blk, 0, stream>>>(vi8, wvi8, bv, Vp, Mrows, Dm, Dm,
                                         scales, 2, 5, scales + 9);

  // 5) quantize projected activations (per-tensor, same elements as heads view)
  qa_quantize<<<1024, blk, 0, stream>>>(Qp, Qpi8, nAct, scales + 7);
  qa_quantize<<<1024, blk, 0, stream>>>(Kp, Kpi8, nAct, scales + 8);
  qa_quantize<<<1024, blk, 0, stream>>>(Vp, Vpi8, nAct, scales + 9);

  // 6-7) attention: stats pass + PV pass
  dim3 gAttn(Bb * Hh, Nseq / 128);
  qa_attn_stats<<<gAttn, blk, 0, stream>>>(Qpi8, Kpi8, scales, rowMv, rowZv,
                                           scales + 11);
  qa_attn_pv<<<gAttn, blk, 0, stream>>>(Qpi8, Kpi8, Vpi8, scales, rowMv, rowZv,
                                        Att, scales + 10);

  // 8) output projection
  qa_quantize<<<1024, blk, 0, stream>>>(Att, atti8, nAct, scales + 10);
  qa_gemm_iu8<<<gGemm, blk, 0, stream>>>(atti8, woi8, bo, out, Mrows, Dm, Dm,
                                         scales, 10, 6, nullptr);
}
